// GewaNet_25271587570245
// MI455X (gfx1250) — matrix operand order repaired and checked by reading, not compiled, not run on hardware
//
#include <hip/hip_runtime.h>
#include <hip/hip_bf16.h>

#define NPTS  4096
#define BATCH 4
#define KNN   16
#define NEDGE (BATCH * NPTS * KNN)   // 262144
#define NPT   (BATCH * NPTS)         // 16384

typedef __attribute__((ext_vector_type(16))) _Float16 v16h;
typedef __attribute__((ext_vector_type(8)))  _Float16 v8h;
typedef __attribute__((ext_vector_type(8)))  float    v8f;

// ---------------------------------------------------------------------------
// helpers
// ---------------------------------------------------------------------------

// float atomic-max via monotone integer trick (matches segment_max semantics)
__device__ __forceinline__ void atomicMaxF(float* addr, float val) {
    if (val >= 0.0f) atomicMax((int*)addr, __float_as_int(val));
    else             atomicMin((unsigned int*)addr, __float_as_uint(val));
}

// A-fragment (16x32 f16) from an LDS tile (row-major, `ld` halfs per row).
// lane&15 = row M; per-lane K values are 16 contiguous halfs
// {kh..kh+15}, kh = (lane>>4)*16 (matches gblB).
__device__ __forceinline__ v16h ldsA(const _Float16* tile, int ld, int lane, int k0) {
    const int row = lane & 15;
    const int kh  = (lane >> 4) * 16;
    const _Float16* p = tile + row * ld + k0;
    v8h g0 = *(const v8h*)(p + kh);
    v8h g1 = *(const v8h*)(p + 8 + kh);
    v16h a;
#pragma unroll
    for (int i = 0; i < 8; ++i) { a[i] = g0[i]; a[8 + i] = g1[i]; }
    return a;
}

// B-fragment (32x16 f16) from transposed f16 weights Wt[O][Ipad]:
// lane&15 = column N, K = kt*32 + (lane>>4)*16 + e  -> 16 contiguous halfs.
__device__ __forceinline__ v16h gblB(const _Float16* Wt, int Ipad, int lane, int nt, int kt) {
    const int n  = nt * 16 + (lane & 15);
    const int kb = kt * 32 + (lane >> 4) * 16;
    return *(const v16h*)(Wt + (long)n * Ipad + kb);
}

// 16-row x 64-col register-blocked GEMM panel: 4 accumulators share each
// A fragment (4x LDS traffic reduction, clause-able B loads, and the 4
// independent WMMAs cover the WMMA->WMMA RAW hazard window).
__device__ __forceinline__ void panel4(const _Float16* Atile, int lda,
                                       const _Float16* Wt, int Ipad,
                                       int lane, int ng, int KT, v8f acc[4]) {
#pragma unroll 1
    for (int kt = 0; kt < KT; ++kt) {
        v16h a = ldsA(Atile, lda, lane, kt * 32);
#pragma unroll
        for (int j = 0; j < 4; ++j) {
            v16h bf = gblB(Wt, Ipad, lane, ng * 4 + j, kt);
            acc[j] = __builtin_amdgcn_wmma_f32_16x16x32_f16(false, a, false, bf,
                                                            (short)0, acc[j], false, false);
        }
    }
}

// BN (+bias) + ReLU epilogue, f32 acc -> f16 LDS activation tile
__device__ __forceinline__ void epiBnRelu(const v8f acc[4], _Float16* out, int ldo,
                                          int lane, int ng,
                                          const float* __restrict__ bias,
                                          const float* __restrict__ gamma,
                                          const float* __restrict__ beta) {
    const int col0 = lane & 15;
    const int rB   = (lane >> 4) * 8;
#pragma unroll
    for (int j = 0; j < 4; ++j) {
        int col = (ng * 4 + j) * 16 + col0;
        float g = gamma[col], bb = bias[col], s = beta[col];
#pragma unroll
        for (int r = 0; r < 8; ++r) {
            float v = (acc[j][r] + bb) * g + s;
            out[(rB + r) * ldo + col] = (_Float16)(v > 0.0f ? v : 0.0f);
        }
    }
}

// distributed cacheline prefetch of a weight matrix (global_prefetch_b8)
__device__ __forceinline__ void prefetchW(const _Float16* Wt, int halfs,
                                          int tid, int nthreads) {
    const char* p = (const char*)Wt;
    for (int o = tid * 128; o < halfs * 2; o += nthreads * 128)
        __builtin_prefetch(p + o, 0, 1);
}

// ---------------------------------------------------------------------------
// weight prep: fp32 W[I][O] -> f16 transposed, K-padded Wt[O][Ipad]
// ---------------------------------------------------------------------------
__global__ void wprep_kernel(const float* __restrict__ W, _Float16* __restrict__ Wt,
                             int I, int O, int Ipad) {
    int i = blockIdx.x * blockDim.x + threadIdx.x;
    if (i >= O * Ipad) return;
    int o = i / Ipad, k = i % Ipad;
    Wt[i] = (k < I) ? (_Float16)W[(long)k * O + o] : (_Float16)0.0f;
}

__global__ void init_neginf_kernel(float* __restrict__ p, int n) {
    int i = blockIdx.x * blockDim.x + threadIdx.x;
    if (i < n) p[i] = __int_as_float(0xFF800000);  // -inf
}

// ---------------------------------------------------------------------------
// kNN: one query point per thread; candidates tiled through LDS.
// All 128 queries in a block share the same batch (4096 % 128 == 0).
// ---------------------------------------------------------------------------
__global__ __launch_bounds__(128) void knn_kernel(const float* __restrict__ pos,
                                                  int* __restrict__ knn) {
    __shared__ float sp[128 * 3];
    const int tid = threadIdx.x;
    const int q   = blockIdx.x * 128 + tid;
    const int b   = q / NPTS;
    const int n   = q % NPTS;
    const float px = pos[((long)b * NPTS + n) * 3 + 0];
    const float py = pos[((long)b * NPTS + n) * 3 + 1];
    const float pz = pos[((long)b * NPTS + n) * 3 + 2];

    float bd[KNN]; int bi[KNN];
#pragma unroll
    for (int i = 0; i < KNN; ++i) { bd[i] = 3.4e38f; bi[i] = 0; }

    for (int c0 = 0; c0 < NPTS; c0 += 128) {
        for (int i = tid; i < 128 * 3; i += 128)
            sp[i] = pos[((long)b * NPTS + c0) * 3 + i];
        __syncthreads();
#pragma unroll 4
        for (int j = 0; j < 128; ++j) {
            float dx = px - sp[j * 3 + 0];
            float dy = py - sp[j * 3 + 1];
            float dz = pz - sp[j * 3 + 2];
            float d  = dx * dx + dy * dy + dz * dz;
            if (d < bd[KNN - 1]) {                 // sorted-insert into top-16
                int s = KNN - 1;
                while (s > 0 && bd[s - 1] > d) { bd[s] = bd[s - 1]; bi[s] = bi[s - 1]; --s; }
                bd[s] = d; bi[s] = c0 + j;
            }
        }
        __syncthreads();
    }
#pragma unroll
    for (int k = 0; k < KNN; ++k) knn[(long)q * KNN + k] = bi[k];
}

// ---------------------------------------------------------------------------
// Fused PointNetConv edge MLP 6(pad32)->64->128->256 + scatter-max.
// 64 edges / block, 4 waves, each wave owns 16 rows (one WMMA M-tile).
// ---------------------------------------------------------------------------
__global__ __launch_bounds__(128) void conv_fused_kernel(
    const float* __restrict__ pos, const int* __restrict__ knn,
    const _Float16* __restrict__ cw1t, const _Float16* __restrict__ cw2t,
    const _Float16* __restrict__ cw3t,
    const float* __restrict__ cb1, const float* __restrict__ cg1, const float* __restrict__ cs1,
    const float* __restrict__ cb2, const float* __restrict__ cg2, const float* __restrict__ cs2,
    const float* __restrict__ cb3,
    float* __restrict__ x) {
    __shared__ __align__(64) _Float16 a0[64 * 32];
    __shared__ __align__(64) _Float16 h1[64 * 64];
    __shared__ __align__(64) _Float16 h2[64 * 128];

    const int  tid = threadIdx.x;
    const long e0  = (long)blockIdx.x * 64;

    // warm the weight path while we gather positions
    prefetchW(cw1t, 64 * 32, tid, 128);
    prefetchW(cw2t, 128 * 64, tid, 128);
    prefetchW(cw3t, 256 * 128, tid, 128);

    // build padded edge features: [pos_src, pos_src - pos_tgt, 0...]
    for (int i = tid; i < 64 * 32; i += 128) {
        int r = i >> 5, c = i & 31;
        long e  = e0 + r;
        int  b  = (int)(e / (NPTS * KNN));
        int  rm = (int)(e - (long)b * NPTS * KNN);
        int  n  = rm / KNN;
        float v = 0.0f;
        if (c < 3) {
            v = pos[((long)b * NPTS + n) * 3 + c];
        } else if (c < 6) {
            int tgt = knn[e];
            v = pos[((long)b * NPTS + n) * 3 + (c - 3)] -
                pos[((long)b * NPTS + tgt) * 3 + (c - 3)];
        }
        a0[i] = (_Float16)v;
    }
    __syncthreads();

    const int wave = tid >> 5;
    const int lane = tid & 31;
    const int col0 = lane & 15;
    const int rB   = (lane >> 4) * 8;          // D-fragment row base
    const _Float16* aT0 = a0 + wave * 16 * 32;
    _Float16*       h1T = h1 + wave * 16 * 64;
    _Float16*       h2T = h2 + wave * 16 * 128;

    // ---- layer 1: 32 -> 64 (1 k-tile, one 4-wide panel) ----
    {
        v8f acc[4] = {};
        panel4(aT0, 32, cw1t, 32, lane, 0, 1, acc);
        epiBnRelu(acc, h1T, 64, lane, 0, cb1, cg1, cs1);
    }
    __syncthreads();

    // ---- layer 2: 64 -> 128 (2 k-tiles, 2 panels) ----
#pragma unroll 1
    for (int ng = 0; ng < 2; ++ng) {
        v8f acc[4] = {};
        panel4(h1T, 64, cw2t, 64, lane, ng, 2, acc);
        epiBnRelu(acc, h2T, 128, lane, ng, cb2, cg2, cs2);
    }
    __syncthreads();

    // ---- layer 3: 128 -> 256 (4 k-tiles, 4 panels), scatter-max ----
    long nb[8];
#pragma unroll
    for (int r = 0; r < 8; ++r) {
        long e  = e0 + wave * 16 + rB + r;
        int  b  = (int)(e / (NPTS * KNN));
        int  tg = knn[e];
        nb[r] = ((long)b * NPTS + tg) * 256;
    }
#pragma unroll 1
    for (int ng = 0; ng < 4; ++ng) {
        v8f acc[4] = {};
        panel4(h2T, 128, cw3t, 128, lane, ng, 4, acc);
#pragma unroll
        for (int j = 0; j < 4; ++j) {
            int col  = (ng * 4 + j) * 16 + col0;
            float bb = cb3[col];
#pragma unroll
            for (int r = 0; r < 8; ++r)
                atomicMaxF(&x[nb[r] + col], acc[j][r] + bb);
        }
    }
}

// ---------------------------------------------------------------------------
// Fused GlobalSA MLP 259(pad288)->256->512->1024 + global max pool.
// 16 points / block, single wave.
// ---------------------------------------------------------------------------
__global__ __launch_bounds__(32) void sa_fused_kernel(
    const float* __restrict__ pos, const float* __restrict__ x,
    const _Float16* __restrict__ sw1t, const _Float16* __restrict__ sw2t,
    const _Float16* __restrict__ sw3t,
    const float* __restrict__ sb1, const float* __restrict__ sg1, const float* __restrict__ ss1,
    const float* __restrict__ sb2, const float* __restrict__ sg2, const float* __restrict__ ss2,
    const float* __restrict__ sb3,
    float* __restrict__ scene) {
    __shared__ __align__(64) _Float16 a0[16 * 288];
    __shared__ __align__(64) _Float16 h1[16 * 256];
    __shared__ __align__(64) _Float16 h2[16 * 512];

    const int tid = threadIdx.x;
    const int p0  = blockIdx.x * 16;

    prefetchW(sw1t, 256 * 288, tid, 32);

    for (int i = tid; i < 16 * 288; i += 32) {
        int r = i / 288, c = i % 288;
        int pt = p0 + r;
        float v = 0.0f;
        if (c < 256)      v = x[(long)pt * 256 + c];
        else if (c < 259) v = pos[(long)pt * 3 + (c - 256)];
        a0[i] = (_Float16)v;
    }
    __syncthreads();

    const int lane = tid;
    const int col0 = lane & 15;
    const int rB   = (lane >> 4) * 8;

    // ---- layer 1: 288 -> 256 (9 k-tiles, 4 panels) ----
    prefetchW(sw2t, 512 * 256, tid, 32);
#pragma unroll 1
    for (int ng = 0; ng < 4; ++ng) {
        v8f acc[4] = {};
        panel4(a0, 288, sw1t, 288, lane, ng, 9, acc);
        epiBnRelu(acc, h1, 256, lane, ng, sb1, sg1, ss1);
    }
    __syncthreads();

    // ---- layer 2: 256 -> 512 (8 k-tiles, 8 panels) ----
    prefetchW(sw3t, 1024 * 512, tid, 32);
#pragma unroll 1
    for (int ng = 0; ng < 8; ++ng) {
        v8f acc[4] = {};
        panel4(h1, 256, sw2t, 256, lane, ng, 8, acc);
        epiBnRelu(acc, h2, 512, lane, ng, sb2, sg2, ss2);
    }
    __syncthreads();

    // ---- layer 3: 512 -> 1024 (16 k-tiles, 16 panels), fused max pool ----
    int bidx[8];
#pragma unroll
    for (int r = 0; r < 8; ++r) bidx[r] = (p0 + rB + r) / NPTS;
#pragma unroll 1
    for (int ng = 0; ng < 16; ++ng) {
        v8f acc[4] = {};
        panel4(h2, 512, sw3t, 512, lane, ng, 16, acc);
#pragma unroll
        for (int j = 0; j < 4; ++j) {
            int col  = (ng * 4 + j) * 16 + col0;
            float bb = sb3[col];
#pragma unroll
            for (int r = 0; r < 8; ++r)
                atomicMaxF(&scene[(long)bidx[r] * 1024 + col], acc[j][r] + bb);
        }
    }
}

// ---------------------------------------------------------------------------
// Predictor MLP 1280->512->256->16 for 4 rows — scalar (far below a WMMA tile)
// ---------------------------------------------------------------------------
__global__ __launch_bounds__(512) void predictor_kernel(
    const float* __restrict__ x, const float* __restrict__ scene,
    const int* __restrict__ qidx,
    const float* __restrict__ pw1, const float* __restrict__ pb1,
    const float* __restrict__ pw2, const float* __restrict__ pb2,
    const float* __restrict__ pw3, const float* __restrict__ pb3,
    float* __restrict__ out) {
    __shared__ float ef[4 * 1280];
    __shared__ float h1s[4 * 512];
    __shared__ float h2s[4 * 256];
    const int tid = threadIdx.x;

    for (int i = tid; i < 4 * 1280; i += 512) {
        int b = i / 1280, c = i % 1280;
        ef[i] = (c < 256) ? x[((long)b * NPTS + qidx[b]) * 256 + c]
                          : scene[(long)b * 1024 + (c - 256)];
    }
    __syncthreads();

    for (int b = 0; b < 4; ++b) {
        float acc = pb1[tid];
        for (int c = 0; c < 1280; ++c) acc += ef[b * 1280 + c] * pw1[(long)c * 512 + tid];
        h1s[b * 512 + tid] = acc > 0.0f ? acc : 0.0f;
    }
    __syncthreads();

    if (tid < 256) {
        for (int b = 0; b < 4; ++b) {
            float acc = pb2[tid];
            for (int c = 0; c < 512; ++c) acc += h1s[b * 512 + c] * pw2[(long)c * 256 + tid];
            h2s[b * 256 + tid] = acc > 0.0f ? acc : 0.0f;
        }
    }
    __syncthreads();

    if (tid < 16) {
        for (int b = 0; b < 4; ++b) {
            float acc = pb3[tid];
            for (int c = 0; c < 256; ++c) acc += h2s[b * 256 + c] * pw3[(long)c * 16 + tid];
            out[b * 16 + tid] = acc;
        }
    }
}

// ---------------------------------------------------------------------------
// host launch
// ---------------------------------------------------------------------------
extern "C" void kernel_launch(void* const* d_in, const int* in_sizes, int n_in,
                              void* d_out, int out_size, void* d_ws, size_t ws_size,
                              hipStream_t stream) {
    (void)in_sizes; (void)n_in; (void)out_size; (void)ws_size;

    const float* pos  = (const float*)d_in[0];
    const int*   qidx = (const int*)d_in[1];
    const float* cw1 = (const float*)d_in[2];  const float* cb1 = (const float*)d_in[3];
    const float* cg1 = (const float*)d_in[4];  const float* cs1 = (const float*)d_in[5];
    const float* cw2 = (const float*)d_in[6];  const float* cb2 = (const float*)d_in[7];
    const float* cg2 = (const float*)d_in[8];  const float* cs2 = (const float*)d_in[9];
    const float* cw3 = (const float*)d_in[10]; const float* cb3 = (const float*)d_in[11];
    const float* sw1 = (const float*)d_in[12]; const float* sb1 = (const float*)d_in[13];
    const float* sg1 = (const float*)d_in[14]; const float* ss1 = (const float*)d_in[15];
    const float* sw2 = (const float*)d_in[16]; const float* sb2 = (const float*)d_in[17];
    const float* sg2 = (const float*)d_in[18]; const float* ss2 = (const float*)d_in[19];
    const float* sw3 = (const float*)d_in[20]; const float* sb3 = (const float*)d_in[21];
    const float* pw1 = (const float*)d_in[22]; const float* pb1 = (const float*)d_in[23];
    const float* pw2 = (const float*)d_in[24]; const float* pb2 = (const float*)d_in[25];
    const float* pw3 = (const float*)d_in[26]; const float* pb3 = (const float*)d_in[27];

    // workspace carve-up (256B aligned)
    char*  ws  = (char*)d_ws;
    size_t off = 0;
    auto carve = [&](size_t bytes) -> char* {
        char* p = ws + off;
        off = (off + bytes + 255) & ~(size_t)255;
        return p;
    };
    int*       knn   = (int*)      carve((size_t)NEDGE * 4);
    float*     x     = (float*)    carve((size_t)NPT * 256 * 4);
    float*     scene = (float*)    carve((size_t)BATCH * 1024 * 4);
    _Float16*  cw1t  = (_Float16*) carve((size_t)64 * 32 * 2);
    _Float16*  cw2t  = (_Float16*) carve((size_t)128 * 64 * 2);
    _Float16*  cw3t  = (_Float16*) carve((size_t)256 * 128 * 2);
    _Float16*  sw1t  = (_Float16*) carve((size_t)256 * 288 * 2);
    _Float16*  sw2t  = (_Float16*) carve((size_t)512 * 256 * 2);
    _Float16*  sw3t  = (_Float16*) carve((size_t)1024 * 512 * 2);

    auto prep = [&](const float* W, _Float16* Wt, int I, int O, int Ipad) {
        int n = O * Ipad;
        wprep_kernel<<<(n + 255) / 256, 256, 0, stream>>>(W, Wt, I, O, Ipad);
    };
    prep(cw1, cw1t,   6,   64,  32);
    prep(cw2, cw2t,  64,  128,  64);
    prep(cw3, cw3t, 128,  256, 128);
    prep(sw1, sw1t, 259,  256, 288);
    prep(sw2, sw2t, 256,  512, 256);
    prep(sw3, sw3t, 512, 1024, 512);

    init_neginf_kernel<<<(NPT * 256 + 255) / 256, 256, 0, stream>>>(x, NPT * 256);
    init_neginf_kernel<<<(BATCH * 1024 + 255) / 256, 256, 0, stream>>>(scene, BATCH * 1024);

    knn_kernel<<<NPT / 128, 128, 0, stream>>>(pos, knn);

    conv_fused_kernel<<<NEDGE / 64, 128, 0, stream>>>(
        pos, knn, cw1t, cw2t, cw3t,
        cb1, cg1, cs1, cb2, cg2, cs2, cb3, x);

    sa_fused_kernel<<<NPT / 16, 32, 0, stream>>>(
        pos, x, sw1t, sw2t, sw3t,
        sb1, sg1, ss1, sb2, sg2, ss2, sb3, scene);

    predictor_kernel<<<1, 512, 0, stream>>>(
        x, scene, qidx, pw1, pb1, pw2, pb2, pw3, pb3, (float*)d_out);
}
